// GlobalSkipFFN_77343771066815
// MI455X (gfx1250) — matrix operand order repaired and checked by reading, not compiled, not run on hardware
//
#include <hip/hip_runtime.h>
#include <hip/hip_bf16.h>

typedef __bf16 bf16;
typedef __attribute__((ext_vector_type(16))) __bf16 v16bf;
typedef __attribute__((ext_vector_type(4)))  __bf16 v4bf;
typedef __attribute__((ext_vector_type(8)))  float  v8f;

// ---------------------------------------------------------------------------
// f32 -> bf16 conversion (vectorized: float4 in, 4x bf16 out)
// ---------------------------------------------------------------------------
__global__ __launch_bounds__(256)
void cvt_f32_bf16(const float* __restrict__ in, bf16* __restrict__ out, int n4) {
    int i = blockIdx.x * blockDim.x + threadIdx.x;
    if (i >= n4) return;
    float4 f = reinterpret_cast<const float4*>(in)[i];
    v4bf o;
    o[0] = (bf16)f.x; o[1] = (bf16)f.y; o[2] = (bf16)f.z; o[3] = (bf16)f.w;
    reinterpret_cast<v4bf*>(out)[i] = o;
}

// ---------------------------------------------------------------------------
// Tiled bf16 WMMA GEMM:  C[M,N] = A[M,K] * B[N,K]^T   (both row-major, bf16)
// Block tile 128x256, BK=64, 8 waves (wave32), wave tile 64(M) x 64(N):
// 4x4 f32 accumulators per wave -> 32 WMMAs per barrier pair.
// GELU=1: C = (bf16) gelu(acc) ; GELU=0: C = (f32) acc
// ---------------------------------------------------------------------------
template <int GELU>
__global__ __launch_bounds__(256)
void wmma_gemm(const bf16* __restrict__ A, const bf16* __restrict__ B,
               float* __restrict__ Cf, bf16* __restrict__ Cb,
               int M, int N, int K) {
    constexpr int BM = 128, BN = 256, BK = 64;
    constexpr int LDT = BK + 8;              // padded LDS row stride (bf16 units)

    __shared__ bf16 sA[BM * LDT];            // 18 KB
    __shared__ bf16 sB[BN * LDT];            // 36 KB

    const int tid   = threadIdx.x;
    const int lane  = tid & 31;
    const int wid   = tid >> 5;
    const int waveM = wid & 1;               // 2 waves along M (64 each)
    const int waveN = wid >> 1;              // 4 waves along N (64 each)
    const int m0 = blockIdx.y * BM;
    const int n0 = blockIdx.x * BN;

    const int fm = lane & 15;                // fragment row/col within 16
    const int kA = (lane < 16) ? 0 : 8;      // A frag: K 0-7/16-23 vs 8-15/24-31
    const int kB = (lane < 16) ? 0 : 8;      // B frag: K 0-7/16-23 vs 8-15/24-31

    const v8f zero = {0.f, 0.f, 0.f, 0.f, 0.f, 0.f, 0.f, 0.f};
    v8f acc[4][4];
#pragma unroll
    for (int i = 0; i < 4; ++i)
#pragma unroll
        for (int j = 0; j < 4; ++j) acc[i][j] = zero;

    for (int k0 = 0; k0 < K; k0 += BK) {
        // ---- stage A 128x64 and B 256x64 bf16 tiles; fully unrolled,
        //      fixed 16B chunks per thread (no predicated loop) ----
#pragma unroll
        for (int t = 0; t < 4; ++t) {        // A: 1024 chunks of 8 bf16
            int c = tid + t * 256;
            int row = c >> 3, cc = (c & 7) * 8;
            *reinterpret_cast<uint4*>(&sA[row * LDT + cc]) =
                *reinterpret_cast<const uint4*>(&A[(m0 + row) * K + k0 + cc]);
        }
#pragma unroll
        for (int t = 0; t < 8; ++t) {        // B: 2048 chunks of 8 bf16
            int c = tid + t * 256;
            int row = c >> 3, cc = (c & 7) * 8;
            *reinterpret_cast<uint4*>(&sB[row * LDT + cc]) =
                *reinterpret_cast<const uint4*>(&B[(n0 + row) * K + k0 + cc]);
        }
        __syncthreads();

        // ---- two 32-deep WMMA k-steps per stage ----
#pragma unroll
        for (int ks = 0; ks < 2; ++ks) {
            const int kk = ks * 32;
            v16bf aF[4], bF[4];
#pragma unroll
            for (int i = 0; i < 4; ++i) {
                int r = waveM * 64 + i * 16 + fm;
                union { uint4 q[2]; v16bf v; } u;
                u.q[0] = *reinterpret_cast<const uint4*>(&sA[r * LDT + kk + kA]);
                u.q[1] = *reinterpret_cast<const uint4*>(&sA[r * LDT + kk + kA + 16]);
                aF[i] = u.v;
            }
#pragma unroll
            for (int j = 0; j < 4; ++j) {
                int r = waveN * 64 + j * 16 + fm;
                union { uint4 q[2]; v16bf v; } u;
                u.q[0] = *reinterpret_cast<const uint4*>(&sB[r * LDT + kk + kB]);
                u.q[1] = *reinterpret_cast<const uint4*>(&sB[r * LDT + kk + kB + 16]);
                bF[j] = u.v;
            }
#pragma unroll
            for (int i = 0; i < 4; ++i)
#pragma unroll
                for (int j = 0; j < 4; ++j)
                    acc[i][j] = __builtin_amdgcn_wmma_f32_16x16x32_bf16(
                        false, aF[i], false, bF[j], (short)0, acc[i][j],
                        false, false);
        }
        __syncthreads();
    }

    // ---- epilogue: C/D layout is lane<16 -> M=vgpr, lane>=16 -> M=8+vgpr ----
    const int rowHalf = (lane < 16) ? 0 : 8;
#pragma unroll
    for (int i = 0; i < 4; ++i) {
        int mBase = m0 + waveM * 64 + i * 16 + rowHalf;
#pragma unroll
        for (int j = 0; j < 4; ++j) {
            int col = n0 + waveN * 64 + j * 16 + fm;
#pragma unroll
            for (int r = 0; r < 8; ++r) {
                float v = acc[i][j][r];
                int row = mBase + r;
                if (GELU) {
                    float g = 0.5f * v * (1.0f + erff(v * 0.70710678118654752f));
                    Cb[row * N + col] = (bf16)g;
                } else {
                    Cf[row * N + col] = v;
                }
            }
        }
    }
}

// ---------------------------------------------------------------------------
// Launch: convert X/Wup/Wdown to bf16, GEMM1 (+gelu, bf16 H), GEMM2 (f32 out)
// ---------------------------------------------------------------------------
extern "C" void kernel_launch(void* const* d_in, const int* in_sizes, int n_in,
                              void* d_out, int out_size, void* d_ws, size_t ws_size,
                              hipStream_t stream) {
    (void)in_sizes; (void)n_in; (void)out_size; (void)ws_size;

    constexpr int T = 8192, D_IN = 6144, D_FF = 4096, D_OUT = 1024;

    const float* X     = (const float*)d_in[0];   // [T, D_IN]
    const float* Wup   = (const float*)d_in[1];   // [D_FF, D_IN] (mask pre-applied)
    const float* Wdown = (const float*)d_in[2];   // [D_OUT, D_FF]
    float*       out   = (float*)d_out;           // [T, D_OUT]

    char* ws = (char*)d_ws;
    bf16* Xb  = (bf16*)(ws);
    bf16* Wub = (bf16*)(ws + (size_t)T * D_IN * 2);
    bf16* Wdb = (bf16*)(ws + (size_t)T * D_IN * 2 + (size_t)D_FF * D_IN * 2);
    bf16* Hb  = (bf16*)(ws + (size_t)T * D_IN * 2 + (size_t)D_FF * D_IN * 2
                           + (size_t)D_OUT * D_FF * 2);

    int nX4 = T * D_IN / 4;
    int nU4 = D_FF * D_IN / 4;
    int nD4 = D_OUT * D_FF / 4;
    cvt_f32_bf16<<<(nX4 + 255) / 256, 256, 0, stream>>>(X, Xb, nX4);
    cvt_f32_bf16<<<(nU4 + 255) / 256, 256, 0, stream>>>(Wup, Wub, nU4);
    cvt_f32_bf16<<<(nD4 + 255) / 256, 256, 0, stream>>>(Wdown, Wdb, nD4);

    dim3 g1(D_FF / 256, T / 128);   // H = gelu(X * Wup^T), bf16
    wmma_gemm<1><<<g1, 256, 0, stream>>>(Xb, Wub, nullptr, Hb, T, D_FF, D_IN);

    dim3 g2(D_OUT / 256, T / 128);  // Y = H * Wdown^T, f32
    wmma_gemm<0><<<g2, 256, 0, stream>>>(Hb, Wdb, out, nullptr, T, D_OUT, D_FF);
}
